// OptimizedSpikingRouter_39865886441622
// MI455X (gfx1250) — matrix operand order repaired and checked by reading, not compiled, not run on hardware
//
#include <hip/hip_runtime.h>

typedef __attribute__((ext_vector_type(4)))  _Float16 v4h;
typedef __attribute__((ext_vector_type(8)))  _Float16 v8h;
typedef __attribute__((ext_vector_type(16))) _Float16 v16h;
typedef __attribute__((ext_vector_type(8)))  float    v8f;

// Problem sizes (fixed by the reference)
#define BB 4096   // batch
#define KK 2048   // input features (GEMM K)
#define HH 4096   // hidden neurons

#define BLK_M 128
#define BLK_N 128
#define BK    64
#define LDSTR 72          // 64 f16 + 8 f16 pad => 144B row stride (16B aligned, bank-spread)
#define NK    (KK / BK)   // 32 K-steps

__global__ __launch_bounds__(256) void lif_wmma_kernel(
    const float* __restrict__ S,     // [B, K] input spikes (0/1)
    const float* __restrict__ MP,    // [B, H] membrane potential
    const int*   __restrict__ RC,    // [B, H] refractory counter
    const float* __restrict__ W,     // [H, K]
    const float* __restrict__ Bias,  // [H]
    float* __restrict__ OutSpk,      // [B, H]
    float* __restrict__ OutV)        // [B, H]
{
    __shared__ alignas(16) _Float16 As[2][BLK_M * LDSTR];   // 2 x 18 KB
    __shared__ alignas(16) _Float16 Bs[2][BLK_N * LDSTR];   // 2 x 18 KB

    const int tid = threadIdx.x;
    const int bm  = blockIdx.y * BLK_M;   // batch-row tile origin
    const int bn  = blockIdx.x * BLK_N;   // neuron-row tile origin

    // cooperative loader: 16 threads per row, one float4 (16B) each; 8 passes x 16 rows
    const int ld_row   = tid >> 4;         // 0..15
    const int ld_chunk = (tid & 15) * 4;   // 0..60 (floats)

    // wave / lane mapping
    const int lane = tid & 31;
    const int wv   = tid >> 5;             // 0..7
    const int wm   = (wv >> 1) * 32;       // wave M offset: 0/32/64/96
    const int wn   = (wv & 1) * 64;        // wave N offset: 0/64
    const int l16  = lane & 15;
    const int hi   = lane >> 4;            // 0: lanes 0-15, 1: lanes 16-31

    // fixed 32-bit per-lane element offsets; the K advance stays scalar (saddr-friendly)
    unsigned offA[8], offB[8];
    #pragma unroll
    for (int it = 0; it < 8; ++it) {
        const int row = it * 16 + ld_row;
        offA[it] = (unsigned)(bm + row) * KK + ld_chunk;
        offB[it] = (unsigned)(bn + row) * KK + ld_chunk;
    }

    v8f acc[2][4] = {};
    float4 aR[8], bR[8];   // in-flight staging registers (double buffering)

    auto load_regs = [&](int kk) {
        const float* Sk = S + kk;          // uniform scalar base advance
        const float* Wk = W + kk;
        #pragma unroll
        for (int it = 0; it < 8; ++it) {
            aR[it] = *(const float4*)(Sk + offA[it]);
            bR[it] = *(const float4*)(Wk + offB[it]);
        }
    };

    auto store_lds = [&](int buf) {
        #pragma unroll
        for (int it = 0; it < 8; ++it) {
            const int row = it * 16 + ld_row;
            v4h ah, bh;
            ah[0] = (_Float16)aR[it].x; ah[1] = (_Float16)aR[it].y;
            ah[2] = (_Float16)aR[it].z; ah[3] = (_Float16)aR[it].w;
            bh[0] = (_Float16)bR[it].x; bh[1] = (_Float16)bR[it].y;
            bh[2] = (_Float16)bR[it].z; bh[3] = (_Float16)bR[it].w;
            *(v4h*)(&As[buf][row * LDSTR + ld_chunk]) = ah;
            *(v4h*)(&Bs[buf][row * LDSTR + ld_chunk]) = bh;
        }
    };

    auto compute = [&](int buf) {
        #pragma unroll
        for (int s = 0; s < 2; ++s) {      // two K=32 substeps within the staged K=64 tile
            v16h afrag[2], bfrag[4];
            #pragma unroll
            for (int mt = 0; mt < 2; ++mt) {
                // lanes 0-15: K 0-7 / 16-23 ; lanes 16-31: K 8-15 / 24-31 (ISA A layout)
                const _Float16* base = &As[buf][(wm + mt * 16 + l16) * LDSTR + s * 32];
                v8h lo = *(const v8h*)(base + hi * 8);
                v8h h8 = *(const v8h*)(base + 16 + hi * 8);
                afrag[mt] = __builtin_shufflevector(lo, h8,
                    0,1,2,3,4,5,6,7,8,9,10,11,12,13,14,15);
            }
            #pragma unroll
            for (int nt = 0; nt < 4; ++nt) {
                // lanes 0-15: N=l16, K 0-7 / 16-23 ; lanes 16-31: N=l16, K 8-15 / 24-31 (matches A)
                const _Float16* base = &Bs[buf][(wn + nt * 16 + l16) * LDSTR + s * 32];
                v8h lo = *(const v8h*)(base + hi * 8);
                v8h h8 = *(const v8h*)(base + 16 + hi * 8);
                bfrag[nt] = __builtin_shufflevector(lo, h8,
                    0,1,2,3,4,5,6,7,8,9,10,11,12,13,14,15);
            }
            #pragma unroll
            for (int mt = 0; mt < 2; ++mt)
                #pragma unroll
                for (int nt = 0; nt < 4; ++nt)
                    acc[mt][nt] = __builtin_amdgcn_wmma_f32_16x16x32_f16(
                        false, afrag[mt], false, bfrag[nt],
                        (short)0, acc[mt][nt], false, false);
        }
    };

    // -------- double-buffered main loop --------
    load_regs(0);
    store_lds(0);
    __syncthreads();

    for (int t = 0; t < NK; ++t) {
        const int cur = t & 1;
        if (t + 1 < NK) {
            load_regs((t + 1) * BK);       // global loads in flight during WMMAs
            if (t + 2 < NK) {              // long-range L2 hint (global_prefetch_b8)
                __builtin_prefetch(S + (t + 2) * BK + offA[0], 0, 1);
                __builtin_prefetch(W + (t + 2) * BK + offB[0], 0, 1);
            }
        }
        compute(cur);
        if (t + 1 < NK) store_lds(cur ^ 1);
        __syncthreads();
    }

    // -------- fused LIF epilogue on the 16x16 f32 C layout --------
    // VGPR r holds M = r + 8*hi, N = l16 (per ISA C/D layout table)
    const float INV_PI = 0.31830988618379067f;
    #pragma unroll
    for (int mt = 0; mt < 2; ++mt) {
        const int grow0 = bm + wm + mt * 16 + hi * 8;
        #pragma unroll
        for (int nt = 0; nt < 4; ++nt) {
            const int gcol = bn + wn + nt * 16 + l16;
            const float bias = Bias[gcol];
            #pragma unroll
            for (int r = 0; r < 8; ++r) {
                const size_t idx = (size_t)(grow0 + r) * HH + gcol;
                const float mpv = __builtin_nontemporal_load(MP + idx);
                const int   rcv = __builtin_nontemporal_load(RC + idx);
                const float cur = acc[mt][nt][r] + bias;
                float v = mpv * 0.9f + ((rcv == 0) ? cur : 0.0f);
                const bool fired = (v >= 1.0f);
                const float vr = fired ? 0.0f : v;          // reset
                const float x  = 2.0f * (vr - 1.0f);        // ALPHA * x_norm
                float sg = (2.0f * INV_PI) / (1.0f + x * x);
                sg = fminf(fmaxf(sg, 0.0f), 1.0f);
                const float spk = fired ? sg : 0.0f;        // spikes * surrogate grad
                __builtin_nontemporal_store(spk, OutSpk + idx);
                __builtin_nontemporal_store(vr,  OutV + idx);
            }
        }
    }
}

extern "C" void kernel_launch(void* const* d_in, const int* in_sizes, int n_in,
                              void* d_out, int out_size, void* d_ws, size_t ws_size,
                              hipStream_t stream) {
    (void)in_sizes; (void)n_in; (void)out_size; (void)d_ws; (void)ws_size;
    const float* S    = (const float*)d_in[0];   // input_spikes [B,I]
    const float* MP   = (const float*)d_in[1];   // membrane_potential [B,H]
    const int*   RC   = (const int*)d_in[2];     // refractory_counter [B,H]
    const float* W    = (const float*)d_in[3];   // W [H,I]
    const float* Bias = (const float*)d_in[4];   // b [H]

    float* OutSpk = (float*)d_out;                       // out_spikes [B,H]
    float* OutV   = (float*)d_out + (size_t)BB * HH;     // v_reset    [B,H]

    dim3 grid(HH / BLK_N, BB / BLK_M);   // (32, 32)
    dim3 block(256);
    lif_wmma_kernel<<<grid, block, 0, stream>>>(S, MP, RC, W, Bias, OutSpk, OutV);
}
